// MLPEquivariantDecoder_79259326481074
// MI455X (gfx1250) — matrix operand order repaired and checked by reading, not compiled, not run on hardware
//
#include <hip/hip_runtime.h>
#include <hip/hip_bf16.h>

// ---------------------------------------------------------------------------
// MLP edge decoder on gfx1250 via transposed WMMA formulation.
//   Hᵀ = W1ᵀ · Xᵀ  (A = weight fragments, B = 16-edge activation tile)
// D accumulator (lane=edge col, vgpr=hidden row) feeds the next layer's B
// fragment with a single half-wave __shfl_xor — no LDS round trip.
// ---------------------------------------------------------------------------

typedef __attribute__((ext_vector_type(16))) _Float16 v16h;
typedef __attribute__((ext_vector_type(8)))  _Float16 v8h;
typedef __attribute__((ext_vector_type(8)))  float    v8f;
typedef __attribute__((ext_vector_type(4)))  float    v4f;
typedef __attribute__((ext_vector_type(4)))  unsigned v4u;
typedef __attribute__((ext_vector_type(8)))  unsigned v8u;

#define WMMA_F16(A, B, C) \
  __builtin_amdgcn_wmma_f32_16x16x32_f16(false, (A), false, (B), (short)0, (C), false, false)

// Fast SiLU.  gfx1250 has V_TANH_F32: silu(x) = fma(hx, tanh(hx), hx), hx=x/2
// -> v_mul + v_tanh + v_fma (3 ops, 1 transcendental; TRANS co-executes with WMMA).
__device__ __forceinline__ float silu_f(float x) {
#if __has_builtin(__builtin_amdgcn_tanhf)
  float hx = 0.5f * x;
  return __builtin_fmaf(hx, __builtin_amdgcn_tanhf(hx), hx);
#else
  return x * __builtin_amdgcn_rcpf(1.0f + __expf(-x));
#endif
}

// ---------------------------------------------------------------------------
// Prep: swizzle fp32 weights/biases into WMMA fragment layout in d_ws.
//
// Weight A-fragment (16x32 f16, ISA 7.12.2 "16-bit A-Matrix"):
//   lane L (l=L&15, h=L>>4) holds row M = t*16 + l
//   element e -> K = c*32 + 16h + e (contiguous, same order as B);  A[m][k] = W[k][m]
// Bias C-fragment (32-bit C/D 16x16): lane L vgpr r holds M = t*16 + r + 8h.
//
// 18 weight frags/net: L1 {t=0..3, c=0..1}, L2 same, L3 {t=0, c=0..1}
//  9 bias  frags/net: L1 t=0..3, L2 t=0..3, L3 t=0
// ---------------------------------------------------------------------------
__global__ void mlp_prep_kernel(const float* __restrict__ W1, const float* __restrict__ B1,
                                const float* __restrict__ W2, const float* __restrict__ B2,
                                const float* __restrict__ W3, const float* __restrict__ B3,
                                _Float16* __restrict__ wfrag, float* __restrict__ bfrag) {
  int tid = blockIdx.x * blockDim.x + threadIdx.x;
  if (tid < 18 * 32) {
    int f = tid >> 5, lane = tid & 31;
    int l = lane & 15, h = lane >> 4;
    int layer, t, c;
    if (f < 8)       { layer = 0; t = f >> 1;        c = f & 1; }
    else if (f < 16) { layer = 1; t = (f - 8) >> 1;  c = (f - 8) & 1; }
    else             { layer = 2; t = 0;             c = f - 16; }
    const float* W = (layer == 0) ? W1 : (layer == 1) ? W2 : W3;
    int dout = (layer == 2) ? 3 : 64;
    int m = t * 16 + l;
    _Float16* dst = wfrag + (f * 32 + lane) * 16;
    for (int e = 0; e < 16; ++e) {
      int k = c * 32 + 16 * h + e;
      float v = (m < dout) ? W[k * dout + m] : 0.0f;
      dst[e] = (_Float16)v;
    }
  } else if (tid < 18 * 32 + 9 * 32) {
    int u = tid - 18 * 32;
    int g = u >> 5, lane = u & 31;
    int h = lane >> 4;
    int layer, t;
    if (g < 4)      { layer = 0; t = g; }
    else if (g < 8) { layer = 1; t = g - 4; }
    else            { layer = 2; t = 0; }
    const float* B = (layer == 0) ? B1 : (layer == 1) ? B2 : B3;
    int dout = (layer == 2) ? 3 : 64;
    float* dst = bfrag + (g * 32 + lane) * 8;
    for (int r = 0; r < 8; ++r) {
      int hid = t * 16 + r + 8 * h;
      dst[r] = (hid < dout) ? B[hid] : 0.0f;
    }
  }
}

// ---------------------------------------------------------------------------
// SiLU + f16-pack one accumulator tile (8 f32 -> 4 dwords of packed f16).
// ---------------------------------------------------------------------------
__device__ __forceinline__ v4u silu_pack(v8f t) {
  v8h p;
#pragma unroll
  for (int r = 0; r < 8; ++r)
    p[r] = (_Float16)silu_f(t[r]);
  return __builtin_bit_cast(v4u, p);
}

// ---------------------------------------------------------------------------
// D -> next-layer-B repack at dword granularity.
// tLo holds hid base+0..15 (vgpr r at lane half h' = hid base + r + 8h'),
// tHi holds hid base+16..31.  Next-layer B frag lane (l,h), element e has
// K = 16h + e:
//   lane (l,0): elems 0..7 = own pLo, elems 8..15 = pLo from lane (l,1)
//   lane (l,1): elems 0..7 = pHi from lane (l,0), elems 8..15 = own pHi
// Each lane sends what its partner needs (send = h ? pLo : pHi) through one
// half-wave __shfl_xor(16) of 4 dwords.
// ---------------------------------------------------------------------------
__device__ __forceinline__ v16h pack_pair(v8f tLo, v8f tHi, int h) {
  v4u pLo = silu_pack(tLo);
  v4u pHi = silu_pack(tHi);
  v4u send, recv;
#pragma unroll
  for (int i = 0; i < 4; ++i)
    send[i] = h ? pLo[i] : pHi[i];
#pragma unroll
  for (int i = 0; i < 4; ++i)
    recv[i] = (unsigned)__shfl_xor((int)send[i], 16, 32);
  v8u cat;
#pragma unroll
  for (int i = 0; i < 4; ++i) {
    cat[i]     = h ? recv[i] : pLo[i];   // low  half: K = 16h + 0..7
    cat[i + 4] = h ? pHi[i]  : recv[i];  // high half: K = 16h + 8..15
  }
  return __builtin_bit_cast(v16h, cat);
}

// ---------------------------------------------------------------------------
// Main decoder: one 16-edge tile per wave per iteration, 18 WMMAs per tile.
// ---------------------------------------------------------------------------
__global__ __launch_bounds__(256) void mlp_decoder_kernel(
    const float* __restrict__ coords,
    const float* __restrict__ feat,
    const float* __restrict__ nf,
    const int*   __restrict__ asrc,
    const int*   __restrict__ adst,
    const _Float16* __restrict__ wfrag,
    const float* __restrict__ bfrag,
    float* __restrict__ out,
    int nEdges, int nTiles) {
  const int lane = threadIdx.x & 31;
  const int wave = threadIdx.x >> 5;
  const int l = lane & 15;
  const int h = lane >> 4;

  // All 18 weight A-fragments resident in VGPRs for the whole tile loop.
  v16h wf[18];
#pragma unroll
  for (int i = 0; i < 18; ++i)
    wf[i] = *(const v16h*)(wfrag + (i * 32 + lane) * 16);

  const int wavesPerBlock = blockDim.x >> 5;
  const int waveStride = gridDim.x * wavesPerBlock;

  for (int tile = blockIdx.x * wavesPerBlock + wave; tile < nTiles; tile += waveStride) {
    int edge = tile * 16 + l;
    int eC = (edge < nEdges) ? edge : (nEdges - 1);  // wave stays EXEC-full for WMMA

    // Opaque zero offset (laundered through in-loop asm): defeats LICM of the
    // bias-fragment loads (so they don't pin 72 VGPRs) while *keeping* global
    // address-space provenance -> global_load_b128, not flat_load_b128.
    unsigned boff = 0;
    asm volatile("" : "+s"(boff));
    const float* bf = bfrag + boff;

    // --- First-layer B fragments (K sequential per half-wave: K = 16h+e) ---
    // chunk0 = x[0..31]  = [coord[src] | coord[dst]]  -> h selects src/dst row
    // chunk1 = x[32..63] = feat row                    -> h selects halves
    int node = h ? adst[eC] : asrc[eC];
    const v4f* crow = (const v4f*)(coords + (size_t)node * 16);
    const v4f* frow = (const v4f*)(feat + (size_t)eC * 32 + h * 16);
    v4f c4[4], f4[4];
#pragma unroll
    for (int q = 0; q < 4; ++q) { c4[q] = crow[q]; f4[q] = frow[q]; }
    v16h b0, b1;
#pragma unroll
    for (int q = 0; q < 4; ++q)
#pragma unroll
      for (int j = 0; j < 4; ++j) {
        b0[q * 4 + j] = (_Float16)c4[q][j];
        b1[q * 4 + j] = (_Float16)f4[q][j];
      }

    // Prefetch next tile's feat row (wave-uniform guard).
    {
      int nxtTile = tile + waveStride;
      if (nxtTile < nTiles)
        __builtin_prefetch(feat + (size_t)(nxtTile * 16 + l) * 32 + h * 16, 0, 3);
    }

    // --- Layer 1: H1ᵀ = W1ᵀ Xᵀ + b1 (C initialized with bias fragments) ---
    v8f a0 = *(const v8f*)(bf + (0 * 32 + lane) * 8);
    v8f a1 = *(const v8f*)(bf + (1 * 32 + lane) * 8);
    v8f a2 = *(const v8f*)(bf + (2 * 32 + lane) * 8);
    v8f a3 = *(const v8f*)(bf + (3 * 32 + lane) * 8);
    a0 = WMMA_F16(wf[0], b0, a0); a0 = WMMA_F16(wf[1], b1, a0);
    a1 = WMMA_F16(wf[2], b0, a1); a1 = WMMA_F16(wf[3], b1, a1);
    a2 = WMMA_F16(wf[4], b0, a2); a2 = WMMA_F16(wf[5], b1, a2);
    a3 = WMMA_F16(wf[6], b0, a3); a3 = WMMA_F16(wf[7], b1, a3);

    v16h c0 = pack_pair(a0, a1, h);   // silu(H1) rows 0..31  as B chunk0
    v16h c1 = pack_pair(a2, a3, h);   // silu(H1) rows 32..63 as B chunk1

    // --- Layer 2 ---
    a0 = *(const v8f*)(bf + (4 * 32 + lane) * 8);
    a1 = *(const v8f*)(bf + (5 * 32 + lane) * 8);
    a2 = *(const v8f*)(bf + (6 * 32 + lane) * 8);
    a3 = *(const v8f*)(bf + (7 * 32 + lane) * 8);
    a0 = WMMA_F16(wf[8],  c0, a0); a0 = WMMA_F16(wf[9],  c1, a0);
    a1 = WMMA_F16(wf[10], c0, a1); a1 = WMMA_F16(wf[11], c1, a1);
    a2 = WMMA_F16(wf[12], c0, a2); a2 = WMMA_F16(wf[13], c1, a2);
    a3 = WMMA_F16(wf[14], c0, a3); a3 = WMMA_F16(wf[15], c1, a3);

    v16h d0 = pack_pair(a0, a1, h);
    v16h d1 = pack_pair(a2, a3, h);

    // --- Layer 3 (output padded 3 -> 16 rows) ---
    v8f o = *(const v8f*)(bf + (8 * 32 + lane) * 8);
    o = WMMA_F16(wf[16], d0, o);
    o = WMMA_F16(wf[17], d1, o);

    // Out rows 0..2 live in lanes h==0, vgprs 0..2 (M = r). nf mask + store.
    if (h == 0 && edge < nEdges) {
      float s = nf[edge];
      size_t ob = (size_t)edge * 3;
      out[ob + 0] = o[0] * s;
      out[ob + 1] = o[1] * s;
      out[ob + 2] = o[2] * s;
    }
  }
}

// ---------------------------------------------------------------------------
// Host side
// ---------------------------------------------------------------------------
extern "C" void kernel_launch(void* const* d_in, const int* in_sizes, int n_in,
                              void* d_out, int out_size, void* d_ws, size_t ws_size,
                              hipStream_t stream) {
  const float* coords = (const float*)d_in[0];
  const float* feat_b = (const float*)d_in[1];
  const float* nf_b   = (const float*)d_in[2];
  const float* feat_a = (const float*)d_in[3];
  const float* nf_a   = (const float*)d_in[4];

  const float* PB[6];
  const float* PA[6];
  const int *sb, *db, *sa, *da;

  if (n_in >= 21) {
    // params flattened into 6 leaves per net
    for (int i = 0; i < 6; ++i) {
      PB[i] = (const float*)d_in[5 + i];
      PA[i] = (const float*)d_in[11 + i];
    }
    sb = (const int*)d_in[17]; db = (const int*)d_in[18];
    sa = (const int*)d_in[19]; da = (const int*)d_in[20];
  } else {
    // params concatenated flat per net: W1(64x64) b1(64) W2(64x64) b2(64) W3(64x3) b3(3)
    const float* bb = (const float*)d_in[5];
    const float* ba = (const float*)d_in[6];
    const int off[6] = {0, 4096, 4160, 8256, 8320, 8512};
    for (int i = 0; i < 6; ++i) { PB[i] = bb + off[i]; PA[i] = ba + off[i]; }
    sb = (const int*)d_in[7]; db = (const int*)d_in[8];
    sa = (const int*)d_in[9]; da = (const int*)d_in[10];
  }

  const int EB = in_sizes[2];  // nf_bond count == E_BOND
  const int EA = in_sizes[4];  // nf_ang  count == E_ANG

  // Workspace layout: [wfrag_bond | wfrag_ang | bfrag_bond | bfrag_ang]
  _Float16* wfB = (_Float16*)d_ws;
  _Float16* wfA = wfB + 18 * 32 * 16;
  float* bfB = (float*)(wfA + 18 * 32 * 16);   // offset 36864 B (32B aligned)
  float* bfA = bfB + 9 * 32 * 8;
  (void)ws_size; (void)out_size;

  mlp_prep_kernel<<<4, 256, 0, stream>>>(PB[0], PB[1], PB[2], PB[3], PB[4], PB[5], wfB, bfB);
  mlp_prep_kernel<<<4, 256, 0, stream>>>(PA[0], PA[1], PA[2], PA[3], PA[4], PA[5], wfA, bfA);

  float* outF = (float*)d_out;

  const int TPW = 4;  // target tiles per wave
  {
    int tiles = (EB + 15) / 16;
    int blocks = (tiles + 8 * TPW - 1) / (8 * TPW);
    if (blocks < 1) blocks = 1;
    mlp_decoder_kernel<<<blocks, 256, 0, stream>>>(coords, feat_b, nf_b, sb, db,
                                                   wfB, bfB, outF, EB, tiles);
  }
  {
    int tiles = (EA + 15) / 16;
    int blocks = (tiles + 8 * TPW - 1) / (8 * TPW);
    if (blocks < 1) blocks = 1;
    mlp_decoder_kernel<<<blocks, 256, 0, stream>>>(coords, feat_a, nf_a, sa, da,
                                                   wfA, bfA, outF + (size_t)EB * 3, EA, tiles);
  }
}
